// MLA_43688407335269
// MI455X (gfx1250) — matrix operand order repaired and checked by reading, not compiled, not run on hardware
//
#include <hip/hip_runtime.h>
#include <hip/hip_bf16.h>
#include <math.h>

// ---------------------------------------------------------------------------
// Problem constants (from the reference)
// ---------------------------------------------------------------------------
#define TOK    2048  // T
#define DMODEL 2048  // D
#define NHEAD  16
#define QRANK  1536
#define KVRANK 512
#define NOPE   128
#define ROPE   64
#define VDIM   128
#define QKD    192   // NOPE + ROPE

typedef __attribute__((ext_vector_type(16))) __bf16 v16bf;
typedef __attribute__((ext_vector_type(8)))  float  v8f;

union Frag16 {        // one WMMA 16-bit operand: 8 VGPRs = 32 bytes per lane
    v16bf v;
    uint4 q[2];
};

// fp32 -> bf16, round-half-up; pair-pack via one v_perm_b32
__device__ __forceinline__ __bf16 f2bf(float f) {
    unsigned u = (__float_as_uint(f) + 0x8000u) >> 16;
    return __builtin_bit_cast(__bf16, (unsigned short)u);
}
__device__ __forceinline__ unsigned packbf2(float a, float b) {
    unsigned ua = __float_as_uint(a) + 0x8000u;
    unsigned ub = __float_as_uint(b) + 0x8000u;
    // result: [15:0] = ua[31:16], [31:16] = ub[31:16]
    return __builtin_amdgcn_perm(ub, ua, 0x07060302u);
}

__device__ __forceinline__ v8f wmma_bf16(const v16bf& a, const v16bf& b, const v8f& c) {
    return __builtin_amdgcn_wmma_f32_16x16x32_bf16(
        /*neg_a=*/false, a, /*neg_b=*/false, b,
        /*c_mod=*/(short)0, c, /*reuse_a=*/false, /*reuse_b=*/false);
}

// CDNA5 async global->LDS copy (ASYNCcnt tracked); 16B per lane.
__device__ __forceinline__ void async_g2l_b128(void* lds, const void* g) {
    unsigned           l  = (unsigned)(size_t)lds;          // flat[31:0] == LDS byte addr
    unsigned long long ga = (unsigned long long)(size_t)g;
    asm volatile("global_load_async_to_lds_b128 %0, %1, off"
                 :: "v"(l), "v"(ga) : "memory");
}
__device__ __forceinline__ void wait_async0() {
    asm volatile("s_wait_asynccnt 0x0" ::: "memory");
}

// YaRN inverse frequency (matches _yarn_inv_freq in the reference)
__device__ __forceinline__ float yarn_inv_freq(int j) {   // j in [0,32)
    const float base = 10000.0f, factor = 40.0f;
    float exps = (2.0f * (float)j) / (float)ROPE;
    float fe = __powf(base, -exps);
    float fi = fe / factor;
    const float twoPi = 6.2831853071795864769f;
    float low  = floorf((float)ROPE * __logf(4096.0f / (32.0f * twoPi)) / (2.0f * __logf(base)));
    float high = ceilf ((float)ROPE * __logf(4096.0f / ( 1.0f * twoPi)) / (2.0f * __logf(base)));
    low  = fmaxf(low, 0.0f);
    high = fminf(high, (float)(ROPE - 1));
    float ramp = fminf(fmaxf(((float)j - low) / fmaxf(high - low, 0.001f), 0.0f), 1.0f);
    float mask = 1.0f - ramp;
    return fi * (1.0f - mask) + fe * mask;
}

// ---------------------------------------------------------------------------
// f32 -> bf16 bulk convert (2 elems/thread, packed b32 stores)
// ---------------------------------------------------------------------------
__global__ __launch_bounds__(256) void k_f32_to_bf16(const float* __restrict__ in,
                                                     unsigned* __restrict__ out, size_t n2) {
    size_t i = (size_t)blockIdx.x * blockDim.x + threadIdx.x;
    if (i < n2) out[i] = packbf2(in[2 * i], in[2 * i + 1]);
}

// ---------------------------------------------------------------------------
// GEMM: C[M,N] f32 = A[M,K] bf16 (row major) x B[K,N] f32 weights (row major)
// Block tile 128x64x32, 128 threads (4 waves), wave tile 32x64.
// A tile: async global->LDS copy, double-buffered (ASYNCcnt ping-pong).
// B tile: fp32 -> bf16 + transpose through registers; load early, store late.
// ---------------------------------------------------------------------------
#define BM 128
#define BN 64

__global__ __launch_bounds__(128) void k_gemm_bf16(const __bf16* __restrict__ A,
                                                   const float* __restrict__ B,
                                                   float* __restrict__ C,
                                                   int M, int N, int K) {
    __shared__ __bf16 As[2][BM * 32];   // [row][k]          16 KB
    __shared__ __bf16 Bs[2][BN * 32];   // transposed [n][k]  8 KB

    const int tid  = threadIdx.x;
    const int wave = tid >> 5;
    const int lane = tid & 31;
    const int half = lane >> 4;
    const int rl   = lane & 15;
    const int m0 = blockIdx.y * BM;
    const int n0 = blockIdx.x * BN;

    v8f acc[2][4] = {};
    float4 br0[2], br1[2];

    auto stage_a = [&](int buf, int k0) {     // 512 x 16B async copies
        #pragma unroll
        for (int i = tid; i < 512; i += 128) {
            int row = i >> 2, seg = i & 3;
            async_g2l_b128(&As[buf][row * 32 + seg * 8],
                           A + (size_t)(m0 + row) * K + k0 + seg * 8);
        }
    };
    auto load_b = [&](int k0) {               // 32x64 f32 tile -> regs (k pairs)
        #pragma unroll
        for (int j = 0; j < 2; ++j) {
            int i = tid + j * 128;
            int kp = (i >> 4) << 1, seg = i & 15;
            const float* p = B + (size_t)(k0 + kp) * N + n0 + seg * 4;
            br0[j] = *(const float4*)p;
            br1[j] = *(const float4*)(p + N);
        }
    };
    auto store_b = [&](int buf) {             // convert + transpose, packed b32
        #pragma unroll
        for (int j = 0; j < 2; ++j) {
            int i = tid + j * 128;
            int kp = (i >> 4) << 1, seg = i & 15;
            int nl = seg * 4;
            *(unsigned*)&Bs[buf][(nl + 0) * 32 + kp] = packbf2(br0[j].x, br1[j].x);
            *(unsigned*)&Bs[buf][(nl + 1) * 32 + kp] = packbf2(br0[j].y, br1[j].y);
            *(unsigned*)&Bs[buf][(nl + 2) * 32 + kp] = packbf2(br0[j].z, br1[j].z);
            *(unsigned*)&Bs[buf][(nl + 3) * 32 + kp] = packbf2(br0[j].w, br1[j].w);
        }
    };

    // prologue: stage first K tile
    stage_a(0, 0);
    load_b(0);
    store_b(0);
    wait_async0();
    __syncthreads();

    int buf = 0;
    for (int k0 = 0; k0 < K; k0 += 32) {
        const bool more = (k0 + 32) < K;
        if (more) {                    // overlap next tile with this tile's math
            stage_a(buf ^ 1, k0 + 32);
            load_b(k0 + 32);
        }

        // hoist all fragments, then burst 8 WMMAs under one dscnt wait
        Frag16 fa[2], fb[4];
        #pragma unroll
        for (int s = 0; s < 2; ++s) {
            int arow = wave * 32 + s * 16 + rl;
            fa[s].q[0] = *(const uint4*)(&As[buf][arow * 32 + half * 8]);
            fa[s].q[1] = *(const uint4*)(&As[buf][arow * 32 + 16 + half * 8]);
        }
        #pragma unroll
        for (int nt = 0; nt < 4; ++nt) {
            int bcol = nt * 16 + rl;
            fb[nt].q[0] = *(const uint4*)(&Bs[buf][bcol * 32 + half * 8]);
            fb[nt].q[1] = *(const uint4*)(&Bs[buf][bcol * 32 + 16 + half * 8]);
        }
        #pragma unroll
        for (int s = 0; s < 2; ++s)
            #pragma unroll
            for (int nt = 0; nt < 4; ++nt)
                acc[s][nt] = wmma_bf16(fa[s].v, fb[nt].v, acc[s][nt]);

        if (more) store_b(buf ^ 1);
        wait_async0();
        __syncthreads();
        buf ^= 1;
    }

    #pragma unroll
    for (int s = 0; s < 2; ++s)
        #pragma unroll
        for (int nt = 0; nt < 4; ++nt)
            #pragma unroll
            for (int r = 0; r < 8; ++r) {
                int row = m0 + wave * 32 + s * 16 + half * 8 + r;
                int col = n0 + nt * 16 + rl;
                C[(size_t)row * N + col] = acc[s][nt][r];
            }
}

// ---------------------------------------------------------------------------
// RMSNorm (fp32 in, bf16 out), one row per block
// ---------------------------------------------------------------------------
__global__ __launch_bounds__(256) void k_rmsnorm_bf16(const float* __restrict__ in,
                                                      const float* __restrict__ scale,
                                                      __bf16* __restrict__ out, int R) {
    __shared__ float red[256];
    int t = blockIdx.x, tid = threadIdx.x;
    const float* row = in + (size_t)t * R;
    float ss = 0.f;
    for (int i = tid; i < R; i += 256) { float v = row[i]; ss += v * v; }
    red[tid] = ss;
    __syncthreads();
    for (int s = 128; s > 0; s >>= 1) {
        if (tid < s) red[tid] += red[tid + s];
        __syncthreads();
    }
    float rs = rsqrtf(red[0] / (float)R + 1e-6f);
    for (int i = tid; i < R; i += 256)
        out[(size_t)t * R + i] = f2bf(row[i] * rs * scale[i]);
}

// ---------------------------------------------------------------------------
// q: RoPE on last 64 dims, fold softmax SCALE, repack [T][N][192] -> [N][T][192]
// ---------------------------------------------------------------------------
__global__ __launch_bounds__(192) void k_rope_pack_q(const float* __restrict__ qtnh,
                                                     const int* __restrict__ positions,
                                                     __bf16* __restrict__ Qh, float scl) {
    int t = blockIdx.x, n = blockIdx.y, j = threadIdx.x;
    const float* in = qtnh + ((size_t)t * NHEAD + n) * QKD;
    __bf16* out = Qh + ((size_t)n * TOK + t) * QKD;
    float v;
    if (j < NOPE) {
        v = in[j];
    } else {
        int jj  = j - NOPE;
        int idx = jj & 31;
        float ang = (float)positions[t] * yarn_inv_freq(idx);
        float s, c;
        __sincosf(ang, &s, &c);
        float x1 = in[NOPE + idx], x2 = in[NOPE + 32 + idx];
        v = (jj < 32) ? (x1 * c - x2 * s) : (x2 * c + x1 * s);
    }
    out[j] = f2bf(v * scl);
}

// ---------------------------------------------------------------------------
// kv: RMSNorm first 512 dims -> bf16 latent; RoPE last 64 dims -> bf16 k_rope
// ---------------------------------------------------------------------------
__global__ __launch_bounds__(256) void k_kv_norm_rope(const float* __restrict__ kvpre,
                                                      const float* __restrict__ kscale,
                                                      const int* __restrict__ positions,
                                                      __bf16* __restrict__ kvlat,
                                                      __bf16* __restrict__ krope) {
    __shared__ float red[256];
    int t = blockIdx.x, tid = threadIdx.x;
    const float* in = kvpre + (size_t)t * (KVRANK + ROPE);
    float ss = 0.f;
    for (int i = tid; i < KVRANK; i += 256) { float v = in[i]; ss += v * v; }
    red[tid] = ss;
    __syncthreads();
    for (int s = 128; s > 0; s >>= 1) {
        if (tid < s) red[tid] += red[tid + s];
        __syncthreads();
    }
    float rs = rsqrtf(red[0] / (float)KVRANK + 1e-6f);
    for (int i = tid; i < KVRANK; i += 256)
        kvlat[(size_t)t * KVRANK + i] = f2bf(in[i] * rs * kscale[i]);
    if (tid < ROPE) {
        int idx = tid & 31;
        float ang = (float)positions[t] * yarn_inv_freq(idx);
        float s, c;
        __sincosf(ang, &s, &c);
        float x1 = in[KVRANK + idx], x2 = in[KVRANK + 32 + idx];
        krope[(size_t)t * ROPE + tid] = f2bf(tid < 32 ? x1 * c - x2 * s : x2 * c + x1 * s);
    }
}

// ---------------------------------------------------------------------------
// split kv_up [T][N*256] f32 into K [N][T][192] (nope ++ rope broadcast) and
// V [N][T][128], bf16
// ---------------------------------------------------------------------------
__global__ __launch_bounds__(128) void k_pack_kv(const float* __restrict__ kvup,
                                                 const __bf16* __restrict__ krope,
                                                 __bf16* __restrict__ Kh,
                                                 __bf16* __restrict__ Vh) {
    int t = blockIdx.x, n = blockIdx.y, tid = threadIdx.x;
    const float* in = kvup + (size_t)t * (NHEAD * (NOPE + VDIM)) + n * (NOPE + VDIM);
    Kh[((size_t)n * TOK + t) * QKD + tid]  = f2bf(in[tid]);
    Vh[((size_t)n * TOK + t) * VDIM + tid] = f2bf(in[NOPE + tid]);
    if (tid < ROPE)
        Kh[((size_t)n * TOK + t) * QKD + NOPE + tid] = krope[(size_t)t * ROPE + tid];
}

// ---------------------------------------------------------------------------
// Flash attention. Block = (64 q rows, 1 head), 128 threads = 4 waves,
// each wave owns 16 q rows. Key tiles of 32, causal, online softmax.
// ---------------------------------------------------------------------------
__global__ __launch_bounds__(128) void k_mla_attn(const __bf16* __restrict__ Qh,
                                                  const __bf16* __restrict__ Kh,
                                                  const __bf16* __restrict__ Vh,
                                                  __bf16* __restrict__ O) {
    __shared__ __bf16 vT[VDIM * 32];        // [feature][key_local]
    __shared__ __bf16 Pl[4][16 * 32];       // per-wave probs [row][key_local]

    const int head = blockIdx.y;
    const int tid  = threadIdx.x;
    const int wave = tid >> 5;
    const int lane = tid & 31;
    const int half = lane >> 4;
    const int rl   = lane & 15;
    const int qb   = blockIdx.x * 64 + wave * 16;

    // Q fragments: 16 rows x 192 features, resident for the whole pass
    Frag16 qf[6];
    #pragma unroll
    for (int c = 0; c < 6; ++c) {
        const __bf16* p = Qh + ((size_t)head * TOK + qb + rl) * QKD + c * 32 + half * 8;
        qf[c].q[0] = *(const uint4*)p;
        qf[c].q[1] = *(const uint4*)(p + 16);
    }

    v8f o[8] = {};
    float m[8], l[8];
    #pragma unroll
    for (int r = 0; r < 8; ++r) { m[r] = -INFINITY; l[r] = 0.f; }

    const int ntiles = 2 * (blockIdx.x + 1);

    for (int st = 0; st < ntiles; ++st) {
        const int s0 = st * 32;

        __syncthreads();
        // stage V tile 32x128 transposed; pack key pairs into b32 stores
        #pragma unroll
        for (int i = tid; i < 256; i += 128) {
            int sp  = (i >> 4) << 1;    // even key 0..30
            int seg = i & 15;           // 8-feature segment
            const __bf16* vp = Vh + ((size_t)head * TOK + s0 + sp) * VDIM + seg * 8;
            uint4 d0 = *(const uint4*)vp;
            uint4 d1 = *(const uint4*)(vp + VDIM);
            const unsigned short* u0 = (const unsigned short*)&d0;
            const unsigned short* u1 = (const unsigned short*)&d1;
            #pragma unroll
            for (int j = 0; j < 8; ++j)
                *(unsigned*)&vT[(seg * 8 + j) * 32 + sp] =
                    (unsigned)u0[j] | ((unsigned)u1[j] << 16);
        }
        __syncthreads();

        // scores: 16 rows x 32 keys, contraction over 192
        v8f sc0 = {}, sc1 = {};
        #pragma unroll
        for (int c = 0; c < 6; ++c) {
            Frag16 kb0, kb1;
            const __bf16* kp0 = Kh + ((size_t)head * TOK + s0 + rl) * QKD + c * 32 + half * 8;
            const __bf16* kp1 = kp0 + 16 * QKD;
            kb0.q[0] = *(const uint4*)kp0;  kb0.q[1] = *(const uint4*)(kp0 + 16);
            kb1.q[0] = *(const uint4*)kp1;  kb1.q[1] = *(const uint4*)(kp1 + 16);
            sc0 = wmma_bf16(qf[c].v, kb0.v, sc0);
            sc1 = wmma_bf16(qf[c].v, kb1.v, sc1);
        }

        // causal mask + online softmax; write P (prob) tile to per-wave LDS
        #pragma unroll
        for (int r = 0; r < 8; ++r) {
            int row = qb + half * 8 + r;
            float x0 = (s0 + rl      <= row) ? sc0[r] : -INFINITY;
            float x1 = (s0 + 16 + rl <= row) ? sc1[r] : -INFINITY;
            float tmax = fmaxf(x0, x1);
            #pragma unroll
            for (int off = 1; off < 16; off <<= 1)
                tmax = fmaxf(tmax, __shfl_xor(tmax, off));
            float mnew = fmaxf(m[r], tmax);
            float p0 = 0.f, p1 = 0.f, corr = 1.f;
            if (mnew != -INFINITY) {
                p0 = __expf(x0 - mnew);        // x==-inf -> 0
                p1 = __expf(x1 - mnew);
                corr = __expf(m[r] - mnew);    // m==-inf -> 0
            }
            float ts = p0 + p1;
            #pragma unroll
            for (int off = 1; off < 16; off <<= 1)
                ts += __shfl_xor(ts, off);
            l[r] = l[r] * corr + ts;
            m[r] = mnew;
            #pragma unroll
            for (int nt = 0; nt < 8; ++nt)
                o[nt][r] *= corr;
            Pl[wave][(half * 8 + r) * 32 + rl]      = f2bf(p0);
            Pl[wave][(half * 8 + r) * 32 + 16 + rl] = f2bf(p1);
        }
        __syncthreads();

        // P (16x32 A-layout) x V (32x128) accumulated into o
        Frag16 pf;
        pf.q[0] = *(const uint4*)(&Pl[wave][rl * 32 + half * 8]);
        pf.q[1] = *(const uint4*)(&Pl[wave][rl * 32 + 16 + half * 8]);
        #pragma unroll
        for (int nt = 0; nt < 8; ++nt) {
            Frag16 vf;
            int f = nt * 16 + rl;
            vf.q[0] = *(const uint4*)(&vT[f * 32 + half * 8]);
            vf.q[1] = *(const uint4*)(&vT[f * 32 + 16 + half * 8]);
            o[nt] = wmma_bf16(pf.v, vf.v, o[nt]);
        }
    }

    // epilogue: normalize, pack [T][(head,feature)] bf16 for the W_o GEMM
    #pragma unroll
    for (int r = 0; r < 8; ++r) {
        float inv = 1.f / l[r];
        int row = qb + half * 8 + r;
        #pragma unroll
        for (int nt = 0; nt < 8; ++nt)
            O[(size_t)row * (NHEAD * VDIM) + head * VDIM + nt * 16 + rl] =
                f2bf(o[nt][r] * inv);
    }
}

// ---------------------------------------------------------------------------
// Host launcher
// ---------------------------------------------------------------------------
extern "C" void kernel_launch(void* const* d_in, const int* in_sizes, int n_in,
                              void* d_out, int out_size, void* d_ws, size_t ws_size,
                              hipStream_t stream) {
    (void)in_sizes; (void)n_in; (void)out_size; (void)ws_size;

    const float* x         = (const float*)d_in[0];
    const int*   positions = (const int*)  d_in[1];
    const float* Wqd       = (const float*)d_in[2];
    const float* Wqu       = (const float*)d_in[3];
    const float* Wkvd      = (const float*)d_in[4];
    const float* Wkvu      = (const float*)d_in[5];
    const float* Wo        = (const float*)d_in[6];
    const float* qns       = (const float*)d_in[7];
    const float* kvns      = (const float*)d_in[8];
    float* out = (float*)d_out;
    char*  ws  = (char*)d_ws;

    size_t off = 0;
    auto take = [&](size_t bytes) { size_t o = off; off += (bytes + 255) & ~(size_t)255; return o; };

    __bf16* x_bf  = (__bf16*)(ws + take((size_t)TOK * DMODEL * 2));
    float*  qpre  = (float*) (ws + take((size_t)TOK * QRANK * 4));
    __bf16* qlat  = (__bf16*)(ws + take((size_t)TOK * QRANK * 2));
    float*  qtnh  = (float*) (ws + take((size_t)TOK * NHEAD * QKD * 4));
    __bf16* Qh    = (__bf16*)(ws + take((size_t)NHEAD * TOK * QKD * 2));
    float*  kvpre = (float*) (ws + take((size_t)TOK * (KVRANK + ROPE) * 4));
    __bf16* kvlat = (__bf16*)(ws + take((size_t)TOK * KVRANK * 2));
    __bf16* krope = (__bf16*)(ws + take((size_t)TOK * ROPE * 2));
    float*  kvup  = (float*) (ws + take((size_t)TOK * NHEAD * (NOPE + VDIM) * 4));
    __bf16* Kh    = (__bf16*)(ws + take((size_t)NHEAD * TOK * QKD * 2));
    __bf16* Vh    = (__bf16*)(ws + take((size_t)NHEAD * TOK * VDIM * 2));
    __bf16* attnb = (__bf16*)(ws + take((size_t)TOK * NHEAD * VDIM * 2));

    const float mscale = 0.1f * logf(40.0f) + 1.0f;
    const float scl = powf((float)QKD, -0.5f) * mscale * mscale;   // folded into q

    {
        size_t n2 = (size_t)TOK * DMODEL / 2;
        k_f32_to_bf16<<<dim3((unsigned)((n2 + 255) / 256)), dim3(256), 0, stream>>>(
            x, (unsigned*)x_bf, n2);
    }
    k_gemm_bf16<<<dim3(QRANK / BN, TOK / BM), dim3(128), 0, stream>>>(x_bf, Wqd, qpre, TOK, QRANK, DMODEL);
    k_rmsnorm_bf16<<<dim3(TOK), dim3(256), 0, stream>>>(qpre, qns, qlat, QRANK);
    k_gemm_bf16<<<dim3((NHEAD * QKD) / BN, TOK / BM), dim3(128), 0, stream>>>(qlat, Wqu, qtnh, TOK, NHEAD * QKD, QRANK);
    k_rope_pack_q<<<dim3(TOK, NHEAD), dim3(QKD), 0, stream>>>(qtnh, positions, Qh, scl);
    k_gemm_bf16<<<dim3((KVRANK + ROPE) / BN, TOK / BM), dim3(128), 0, stream>>>(x_bf, Wkvd, kvpre, TOK, KVRANK + ROPE, DMODEL);
    k_kv_norm_rope<<<dim3(TOK), dim3(256), 0, stream>>>(kvpre, kvns, positions, kvlat, krope);
    k_gemm_bf16<<<dim3((NHEAD * (NOPE + VDIM)) / BN, TOK / BM), dim3(128), 0, stream>>>(kvlat, Wkvu, kvup, TOK, NHEAD * (NOPE + VDIM), KVRANK);
    k_pack_kv<<<dim3(TOK, NHEAD), dim3(128), 0, stream>>>(kvup, krope, Kh, Vh);
    k_mla_attn<<<dim3(TOK / 64, NHEAD), dim3(128), 0, stream>>>(Qh, Kh, Vh, attnb);
    k_gemm_bf16<<<dim3(DMODEL / BN, TOK / BM), dim3(128), 0, stream>>>(attnb, Wo, out, TOK, DMODEL, NHEAD * VDIM);
}
